// AUGRU_68659347194009
// MI455X (gfx1250) — matrix operand order repaired and checked by reading, not compiled, not run on hardware
//
#include <hip/hip_runtime.h>

// ---------------------------------------------------------------------------
// AUGRU on gfx1250. Attention path is provably dead (softmax over singleton
// axis == 1), so this is a fused GRU: per-workgroup batch tile of 16 rows,
// all six 128x128 weights resident in LDS as f16 (transposed, padded),
// recurrence driven by v_wmma_f32_16x16x32_f16 with f32 accumulation.
// Ping-pong x/h tiles -> a single workgroup barrier per time step.
// ---------------------------------------------------------------------------

typedef __attribute__((ext_vector_type(16))) _Float16 v16h;
typedef __attribute__((ext_vector_type(8)))  _Float16 v8h;
typedef __attribute__((ext_vector_type(4)))  _Float16 v4h;
typedef __attribute__((ext_vector_type(8)))  float    v8f;

#define B_  1024
#define T_  200
#define E_  128
#define BM  16     // batch rows per workgroup -> 64 workgroups
#define XS  136    // row stride (halfs) for A-side LDS tiles (272B = 17*16, 16B aligned)
#define WS  136    // row stride (halfs) for transposed-weight LDS

// A operand: 16x32 f16 tile, rows m=lane%16, K contiguous (matches load_b):
//   lanes 0-15 : K kb+0..15
//   lanes 16-31: K kb+16..31
__device__ __forceinline__ v16h load_a(const _Float16* base, int lane, int kb) {
    const int m = lane & 15, half = lane >> 4;
    const _Float16* p = base + m * XS + kb + 16 * half;
    v8h c0 = *(const v8h*)(p);        // K = kb + 16*half + 0..7
    v8h c1 = *(const v8h*)(p + 8);    // K = kb + 16*half + 8..15
    return __builtin_shufflevector(c0, c1, 0,1,2,3,4,5,6,7,8,9,10,11,12,13,14,15);
}

// B operand: 32x16 f16 tile from transposed weights Wt[n][k]:
//   lane holds column n=lane%16; lanes 0-15 K=kb..kb+15, lanes 16-31 K=kb+16..kb+31
//   -> 16 contiguous halfs per lane (two ds_load_b128).
__device__ __forceinline__ v16h load_b(const _Float16* wt, int lane, int ntile, int kb) {
    const int n = ntile * 16 + (lane & 15), half = lane >> 4;
    return *(const v16h*)(wt + n * WS + kb + 16 * half);
}

__global__ __launch_bounds__(256) void augru_gfx1250_kernel(
    const float* __restrict__ x,  const float* __restrict__ h0,
    const float* __restrict__ Wu, const float* __restrict__ Uu, const float* __restrict__ bu,
    const float* __restrict__ Wr, const float* __restrict__ Ur, const float* __restrict__ br,
    const float* __restrict__ Wh, const float* __restrict__ Uh, const float* __restrict__ bh,
    float* __restrict__ out)
{
    // ~236 KB static LDS: legal on CDNA5 (320 KB/WGP), one workgroup per WGP.
    __shared__ _Float16 sWu[E_*WS], sUu[E_*WS], sWr[E_*WS],
                        sUr[E_*WS], sWh[E_*WS], sUh[E_*WS];
    __shared__ _Float16 sX[2][BM*XS], sH16[2][BM*XS];   // ping-pong tiles
    __shared__ float    sHF[BM*E_];
    __shared__ float    sBu[E_], sBr[E_], sBh[E_];

    const int tid  = threadIdx.x;
    const int lane = tid & 31;
    const int w    = tid >> 5;          // wave id 0..7 == output-feature ntile
    const int b0   = blockIdx.x * BM;

    // ---- prologue: weights -> LDS (transposed, f16); biases; h0 tile -------
    for (int idx = tid; idx < E_ * E_; idx += 256) {
        const int k = idx >> 7, n = idx & 127;   // W[k][n] -> Wt[n][k]
        sWu[n*WS + k] = (_Float16)Wu[idx];
        sUu[n*WS + k] = (_Float16)Uu[idx];
        sWr[n*WS + k] = (_Float16)Wr[idx];
        sUr[n*WS + k] = (_Float16)Ur[idx];
        sWh[n*WS + k] = (_Float16)Wh[idx];
        sUh[n*WS + k] = (_Float16)Uh[idx];
    }
    if (tid < E_) { sBu[tid] = bu[tid]; sBr[tid] = br[tid]; sBh[tid] = bh[tid]; }
    for (int idx = tid; idx < BM * E_; idx += 256) {
        const int m = idx >> 7, n = idx & 127;
        const float v = h0[(size_t)(b0 + m) * E_ + n];
        sHF[m*E_ + n]       = v;
        sH16[0][m*XS + n]   = (_Float16)v;   // state entering step 0
    }
    __syncthreads();

    // Per-lane output element mapping (C layout): vgpr i -> m = i + 8*half,
    // all vgprs -> feature n = 16*w + (lane%16).
    const int  half  = lane >> 4;
    const int  nfeat = w * 16 + (lane & 15);
    const float bUv = sBu[nfeat], bRv = sBr[nfeat], bHv = sBh[nfeat];

    for (int t = 0; t < T_; ++t) {
        const int pb = t & 1;           // buffer holding h entering step t / x_t

        // ---- stage x_t tile (16 rows x 128 f32 -> f16 LDS), coalesced -----
        // Safe unsynced: laggard waves can only still be reading sX[pb^1].
        for (int idx = tid; idx < BM * (E_ / 4); idx += 256) {
            const int m  = idx >> 5;
            const int c4 = (idx & 31) * 4;
            const size_t gb = ((size_t)(b0 + m) * T_ + t) * E_ + c4;
            const float4 v = *(const float4*)(x + gb);
            v4h hv; hv[0] = (_Float16)v.x; hv[1] = (_Float16)v.y;
                    hv[2] = (_Float16)v.z; hv[3] = (_Float16)v.w;
            *(v4h*)(&sX[pb][m*XS + c4]) = hv;
            if (t + 1 < T_)  // prefetch next step's x tile toward L2/L0
                __builtin_prefetch(x + gb + E_, 0, 1);
        }
        __syncthreads();   // single barrier/step: x_t and h_t (written at end
                           // of step t-1 into buffer pb) are now visible.

        // ---- GEMM stage: 24 WMMAs per wave per step -----------------------
        v8f accU, accR, accXH, accG;
        #pragma unroll
        for (int i = 0; i < 8; ++i) { accU[i] = bUv; accR[i] = bRv; accXH[i] = bHv; accG[i] = 0.f; }

        #pragma unroll
        for (int kt = 0; kt < 4; ++kt) {
            const int kb = kt * 32;
            const v16h ax = load_a(sX[pb],   lane, kb);
            const v16h ah = load_a(sH16[pb], lane, kb);
            accU  = __builtin_amdgcn_wmma_f32_16x16x32_f16(false, ax, false, load_b(sWu, lane, w, kb), (short)0, accU,  false, false);
            accU  = __builtin_amdgcn_wmma_f32_16x16x32_f16(false, ah, false, load_b(sUu, lane, w, kb), (short)0, accU,  false, false);
            accR  = __builtin_amdgcn_wmma_f32_16x16x32_f16(false, ax, false, load_b(sWr, lane, w, kb), (short)0, accR,  false, false);
            accR  = __builtin_amdgcn_wmma_f32_16x16x32_f16(false, ah, false, load_b(sUr, lane, w, kb), (short)0, accR,  false, false);
            accXH = __builtin_amdgcn_wmma_f32_16x16x32_f16(false, ax, false, load_b(sWh, lane, w, kb), (short)0, accXH, false, false);
            accG  = __builtin_amdgcn_wmma_f32_16x16x32_f16(false, ah, false, load_b(sUh, lane, w, kb), (short)0, accG,  false, false);
        }

        // ---- elementwise GRU update; a_t == 1 (dead softmax) --------------
        // Writes go to the *other* buffer; no barrier needed here. sHF
        // elements are private to their owning lane across all steps.
        #pragma unroll
        for (int i = 0; i < 8; ++i) {
            const int m = i + 8 * half;
            const float u  = 1.f / (1.f + __expf(-accU[i]));
            const float r  = 1.f / (1.f + __expf(-accR[i]));
            const float hh = tanhf(accXH[i] + r * accG[i]);
            const float ho = sHF[m*E_ + nfeat];
            const float hn = (1.f - u) * ho + u * hh;
            sHF[m*E_ + nfeat]        = hn;
            sH16[pb ^ 1][m*XS + nfeat] = (_Float16)hn;   // state entering step t+1
            out[((size_t)(b0 + m) * T_ + t) * E_ + nfeat] = hn;
        }
    }

    __syncthreads();   // make all lanes' final sHF writes visible

    // ---- h_last, appended after outs ([B,T,E] then [B,E]) -----------------
    for (int idx = tid; idx < BM * E_; idx += 256) {
        const int m = idx >> 7, n = idx & 127;
        out[(size_t)B_ * T_ * E_ + (size_t)(b0 + m) * E_ + n] = sHF[m*E_ + n];
    }
}

extern "C" void kernel_launch(void* const* d_in, const int* in_sizes, int n_in,
                              void* d_out, int out_size, void* d_ws, size_t ws_size,
                              hipStream_t stream) {
    (void)in_sizes; (void)n_in; (void)out_size; (void)d_ws; (void)ws_size;
    // setup_inputs order: x, item, h0, Wu, Uu, bu, Wr, Ur, br, Wh, Uh, bh, Wa
    // item (1) and Wa (12) are dead (softmax over singleton axis == 1).
    const float* x  = (const float*)d_in[0];
    const float* h0 = (const float*)d_in[2];
    const float* Wu = (const float*)d_in[3];
    const float* Uu = (const float*)d_in[4];
    const float* bu = (const float*)d_in[5];
    const float* Wr = (const float*)d_in[6];
    const float* Ur = (const float*)d_in[7];
    const float* br = (const float*)d_in[8];
    const float* Wh = (const float*)d_in[9];
    const float* Uh = (const float*)d_in[10];
    const float* bh = (const float*)d_in[11];
    float* out = (float*)d_out;

    dim3 grid(B_ / BM);   // 64 workgroups, each owns 16 batch rows for all T
    dim3 block(256);      // 8 wave32s; wave w owns feature tile [16w, 16w+16)
    hipLaunchKernelGGL(augru_gfx1250_kernel, grid, block, 0, stream,
                       x, h0, Wu, Uu, bu, Wr, Ur, br, Wh, Uh, bh, out);
}
